// Seq2SeqGRU_26216480374904
// MI455X (gfx1250) — matrix operand order repaired and checked by reading, not compiled, not run on hardware
//
#include <hip/hip_runtime.h>
#include <hip/hip_bf16.h>

// ---------------- problem constants ----------------
#define VV   32000
#define EE   512
#define HH   1024
#define BB   32
#define SS   64
#define G3H  3072   // 3*H
#define MM   2048   // B*S

typedef __bf16 bf16;
typedef bf16  bfx8   __attribute__((ext_vector_type(8)));
typedef bf16  bfx16  __attribute__((ext_vector_type(16)));
typedef float fx8    __attribute__((ext_vector_type(8)));

// ---------------- WMMA fragment helpers (CDNA5 wave32 layouts) ----------------
// A-matrix 16x32 bf16: lanes 0-15 hold M=0..15 with K runs {8*lg .. 8*lg+7} and
// {16+8*lg .. 16+8*lg+23? no: +7}; lg = lane>>4 selects which K runs.
__device__ __forceinline__ bfx16 load_a_frag(const bf16* __restrict__ A, int lda,
                                             int row_base, int kc, int lane) {
  int m  = lane & 15;
  int lg = lane >> 4;
  const bf16* p = A + (size_t)(row_base + m) * lda + kc + 8 * lg;
  bfx8 lo = *(const bfx8*)p;           // K = kc + 8*lg .. +7        (16B)
  bfx8 hi = *(const bfx8*)(p + 16);    // K = kc + 16 + 8*lg .. +7   (16B)
  return __builtin_shufflevector(lo, hi, 0,1,2,3,4,5,6,7,8,9,10,11,12,13,14,15);
}

// B-matrix 32x16 bf16, B = W^T with W [N,K] row-major: lane holds column
// n = lane&15, K runs {8*lg .. 8*lg+7} and {16+8*lg .. 16+8*lg+7} of W's row.
__device__ __forceinline__ bfx16 load_b_frag(const bf16* __restrict__ W, int ldw,
                                             int col_base, int kc, int lane) {
  int n  = lane & 15;
  int lg = lane >> 4;
  const bf16* p = W + (size_t)(col_base + n) * ldw + kc + 8 * lg;
  bfx8 lo = *(const bfx8*)p;
  bfx8 hi = *(const bfx8*)(p + 16);
  return __builtin_shufflevector(lo, hi, 0,1,2,3,4,5,6,7,8,9,10,11,12,13,14,15);
}

__device__ __forceinline__ fx8 wmma_bf16(bfx16 a, bfx16 b, fx8 c) {
  return __builtin_amdgcn_wmma_f32_16x16x32_bf16(false, a, false, b,
                                                 (short)0, c, false, false);
}

// ---------------- generic GEMM: C[M,N](f32) = A[M,K](bf16) * W[N,K]^T + bias ----------------
// block = 128 threads (4 waves), 64x64 tile; wave -> 32x32 sub-tile (2x2 frags)
__global__ void __launch_bounds__(128)
gemm_bf16_bias(const bf16* __restrict__ A, const bf16* __restrict__ W,
               const float* __restrict__ bias, float* __restrict__ C,
               int M, int N, int K) {
  int lane = threadIdx.x & 31;
  int w    = threadIdx.x >> 5;
  int m0 = blockIdx.y * 64 + (w >> 1) * 32;
  int n0 = blockIdx.x * 64 + (w & 1) * 32;

  fx8 acc[2][2] = {};
  #pragma unroll 4
  for (int kc = 0; kc < K; kc += 32) {
    if (kc + 64 < K) {
      // hint next-next weight chunk into cache (global_prefetch_b8)
      __builtin_prefetch(W + (size_t)(n0 + 16*(lane>>4) + (lane&15)) * K + kc + 64, 0, 1);
    }
    bfx16 a0 = load_a_frag(A, K, m0,      kc, lane);
    bfx16 a1 = load_a_frag(A, K, m0 + 16, kc, lane);
    bfx16 b0 = load_b_frag(W, K, n0,      kc, lane);
    bfx16 b1 = load_b_frag(W, K, n0 + 16, kc, lane);
    acc[0][0] = wmma_bf16(a0, b0, acc[0][0]);
    acc[0][1] = wmma_bf16(a0, b1, acc[0][1]);
    acc[1][0] = wmma_bf16(a1, b0, acc[1][0]);
    acc[1][1] = wmma_bf16(a1, b1, acc[1][1]);
  }

  // C/D layout: n = lane&15 ; m = 8*(lane>>4) + v
  int n_l  = lane & 15;
  int mofs = (lane >> 4) * 8;
  #pragma unroll
  for (int j = 0; j < 2; ++j) {
    int col = n0 + 16*j + n_l;
    float bv = bias ? bias[col] : 0.0f;
    #pragma unroll
    for (int i = 0; i < 2; ++i) {
      int rbase = m0 + 16*i + mofs;
      #pragma unroll
      for (int v = 0; v < 8; ++v)
        C[(size_t)(rbase + v) * N + col] = acc[i][j][v] + bv;
    }
  }
}

// ---------------- fused GRU recurrent step ----------------
// grid = 64 blocks (16 h-columns each), block = 192 threads = 6 waves:
//   wave w: gate = w%3 (r,z,n), K-half = w/3.  Each wave: 2 M-tiles x 16 k-chunks.
// hg partials exchanged via LDS, then fused gate math producing h_new (f32+bf16)
// and y[:,t,:] (bf16 input for the next layer / FC).
__global__ void __launch_bounds__(192)
gru_step(const bf16* __restrict__ h_in_bf, const float* __restrict__ h_in_f,
         const bf16* __restrict__ Whh, const float* __restrict__ bhh,
         const float* __restrict__ xg,
         float* __restrict__ h_out_f, bf16* __restrict__ h_out_bf,
         bf16* __restrict__ y, int t) {
  __shared__ float hg[2][3][32][16];

  int tid  = threadIdx.x;
  int lane = tid & 31;
  int w    = tid >> 5;          // 0..5
  int gate = w % 3;             // r, z, n
  int kh   = w / 3;             // K half (0: 0..511, 1: 512..1023)
  int j0   = blockIdx.x * 16;
  int colb = gate * HH + j0;

  fx8 acc0 = {}, acc1 = {};
  int kbeg = kh * (HH / 2);
  #pragma unroll 4
  for (int kc = kbeg; kc < kbeg + (HH / 2); kc += 32) {
    bfx16 a0 = load_a_frag(h_in_bf, HH, 0,  kc, lane);
    bfx16 a1 = load_a_frag(h_in_bf, HH, 16, kc, lane);
    bfx16 b  = load_b_frag(Whh,     HH, colb, kc, lane);
    acc0 = wmma_bf16(a0, b, acc0);
    acc1 = wmma_bf16(a1, b, acc1);
  }

  int n_l  = lane & 15;
  int mofs = (lane >> 4) * 8;
  #pragma unroll
  for (int v = 0; v < 8; ++v) {
    hg[kh][gate][mofs + v][n_l]      = acc0[v];
    hg[kh][gate][16 + mofs + v][n_l] = acc1[v];
  }
  __syncthreads();

  for (int e = tid; e < BB * 16; e += 192) {
    int m = e >> 4;              // batch row
    int n = e & 15;
    int j = j0 + n;
    const float* xrow = xg + ((size_t)m * SS + t) * G3H;
    float hr = hg[0][0][m][n] + hg[1][0][m][n] + bhh[j];
    float hz = hg[0][1][m][n] + hg[1][1][m][n] + bhh[HH + j];
    float hn = hg[0][2][m][n] + hg[1][2][m][n] + bhh[2 * HH + j];
    float r  = 1.0f / (1.0f + __expf(-(xrow[j] + hr)));
    float z  = 1.0f / (1.0f + __expf(-(xrow[HH + j] + hz)));
    float nn = tanhf(xrow[2 * HH + j] + r * hn);
    float hnew = (1.0f - z) * nn + z * h_in_f[m * HH + j];
    h_out_f[m * HH + j]  = hnew;
    bf16 hb = (bf16)hnew;
    h_out_bf[m * HH + j] = hb;
    y[((size_t)m * SS + t) * HH + j] = hb;
  }
}

// ---------------- elementwise helpers ----------------
__global__ void f32_to_bf16(const float* __restrict__ in, bf16* __restrict__ out, int n) {
  int i = blockIdx.x * blockDim.x + threadIdx.x;
  if (i < n) out[i] = (bf16)in[i];
}

__global__ void embed_gather(const int* __restrict__ X, const float* __restrict__ emb,
                             bf16* __restrict__ out, int rows, int ecols) {
  int i = blockIdx.x * blockDim.x + threadIdx.x;
  if (i >= rows * ecols) return;
  int r = i / ecols;
  int c = i - r * ecols;
  out[i] = (bf16)emb[(size_t)X[r] * ecols + c];
}

// ---------------- host orchestration ----------------
extern "C" void kernel_launch(void* const* d_in, const int* in_sizes, int n_in,
                              void* d_out, int out_size, void* d_ws, size_t ws_size,
                              hipStream_t stream) {
  const int*   enc_X = (const int*)d_in[0];
  const int*   dec_X = (const int*)d_in[1];
  const float* emb_enc = (const float*)d_in[2];
  const float* emb_dec = (const float*)d_in[3];
  const float* Wf[9];   // 8 GRU weights + fc_W (in conversion order)
  Wf[0] = (const float*)d_in[4];   // enc_Wih0
  Wf[1] = (const float*)d_in[5];   // enc_Whh0
  Wf[2] = (const float*)d_in[8];   // enc_Wih1
  Wf[3] = (const float*)d_in[9];   // enc_Whh1
  Wf[4] = (const float*)d_in[12];  // dec_Wih0
  Wf[5] = (const float*)d_in[13];  // dec_Whh0
  Wf[6] = (const float*)d_in[16];  // dec_Wih1
  Wf[7] = (const float*)d_in[17];  // dec_Whh1
  Wf[8] = (const float*)d_in[20];  // fc_W
  const int Wn[9] = { G3H*EE, G3H*HH, G3H*HH, G3H*HH,
                      G3H*EE, G3H*HH, G3H*HH, G3H*HH, VV*HH };
  const float* enc_bih0 = (const float*)d_in[6];
  const float* enc_bhh0 = (const float*)d_in[7];
  const float* enc_bih1 = (const float*)d_in[10];
  const float* enc_bhh1 = (const float*)d_in[11];
  const float* dec_bih0 = (const float*)d_in[14];
  const float* dec_bhh0 = (const float*)d_in[15];
  const float* dec_bih1 = (const float*)d_in[18];
  const float* dec_bhh1 = (const float*)d_in[19];
  const float* fc_b     = (const float*)d_in[21];

  // workspace carve-out (256B aligned)
  char* ws = (char*)d_ws;
  size_t off = 0;
  auto carve = [&](size_t bytes) -> void* {
    off = (off + 255) & ~(size_t)255;
    void* p = ws + off;
    off += bytes;
    return p;
  };
  bf16* Wbf[9];
  for (int i = 0; i < 9; ++i) Wbf[i] = (bf16*)carve((size_t)Wn[i] * sizeof(bf16));
  bf16* x_enc = (bf16*)carve((size_t)MM * EE * sizeof(bf16));
  bf16* x_dec = (bf16*)carve((size_t)MM * EE * sizeof(bf16));
  float* xg   = (float*)carve((size_t)MM * G3H * sizeof(float));
  bf16* yA    = (bf16*)carve((size_t)MM * HH * sizeof(bf16));
  bf16* yB    = (bf16*)carve((size_t)MM * HH * sizeof(bf16));
  float* hF[2][2]; bf16* hB[2][2];
  for (int s = 0; s < 2; ++s)
    for (int b = 0; b < 2; ++b) {
      hF[s][b] = (float*)carve((size_t)BB * HH * sizeof(float));
      hB[s][b] = (bf16*)carve((size_t)BB * HH * sizeof(bf16));
    }
  if (off > ws_size) return;  // insufficient scratch

  // 1) weights -> bf16
  for (int i = 0; i < 9; ++i)
    f32_to_bf16<<<(Wn[i] + 255) / 256, 256, 0, stream>>>(Wf[i], Wbf[i], Wn[i]);

  // 2) embedding gathers -> bf16 activations
  embed_gather<<<(MM * EE + 255) / 256, 256, 0, stream>>>(enc_X, emb_enc, x_enc, MM, EE);
  embed_gather<<<(MM * EE + 255) / 256, 256, 0, stream>>>(dec_X, emb_dec, x_dec, MM, EE);

  // 3) zero initial hidden states (buf 0 of both layer sets)
  for (int s = 0; s < 2; ++s) {
    hipMemsetAsync(hF[s][0], 0, (size_t)BB * HH * sizeof(float), stream);
    hipMemsetAsync(hB[s][0], 0, (size_t)BB * HH * sizeof(bf16), stream);
  }

  // 4) four GRU layers: enc0, enc1, dec0, dec1
  //    (dec layer L starts from enc layer L's final h, which lands in buf 0)
  struct Layer {
    const bf16* x; int K;
    const bf16* wih; const float* bih;
    const bf16* whh; const float* bhh;
    int hset; bf16* yout;
  } layers[4] = {
    { x_enc, EE, Wbf[0], enc_bih0, Wbf[1], enc_bhh0, 0, yA },
    { yA,    HH, Wbf[2], enc_bih1, Wbf[3], enc_bhh1, 1, yB },
    { x_dec, EE, Wbf[4], dec_bih0, Wbf[5], dec_bhh0, 0, yA },
    { yA,    HH, Wbf[6], dec_bih1, Wbf[7], dec_bhh1, 1, yB },
  };

  for (int L = 0; L < 4; ++L) {
    const Layer& ly = layers[L];
    // xg = x @ Wih^T + bih   : [2048, K] x [K, 3072]
    gemm_bf16_bias<<<dim3(G3H / 64, MM / 64), 128, 0, stream>>>(
        ly.x, ly.wih, ly.bih, xg, MM, G3H, ly.K);
    // 64 sequential recurrent steps, double-buffered h (S even -> ends in buf 0)
    for (int t = 0; t < SS; ++t) {
      int in = t & 1, out = in ^ 1;
      gru_step<<<64, 192, 0, stream>>>(
          hB[ly.hset][in], hF[ly.hset][in], ly.whh, ly.bhh, xg,
          hF[ly.hset][out], hB[ly.hset][out], ly.yout, t);
    }
  }

  // 5) logits = d1 @ fc_W^T + fc_b : [2048,1024] x [1024,32000] (L2-resident)
  gemm_bf16_bias<<<dim3(VV / 64, MM / 64), 128, 0, stream>>>(
      yB, Wbf[8], fc_b, (float*)d_out, MM, VV, HH);
}
